// LAGAT_77129022701603
// MI455X (gfx1250) — compile-verified, matrix operand order checked by reading, not run on hardware
//
#include <hip/hip_runtime.h>
#include <math.h>

typedef __attribute__((ext_vector_type(16))) __bf16 v16bf;
typedef __attribute__((ext_vector_type(8)))  __bf16 v8bf;
typedef __attribute__((ext_vector_type(8)))  float  v8f;

__device__ __forceinline__ __bf16 f2bf(float f) {
  unsigned u = __builtin_bit_cast(unsigned, f);
  unsigned r = u + 0x7FFFu + ((u >> 16) & 1u);   // round-to-nearest-even
  unsigned short h = (unsigned short)(r >> 16);
  return __builtin_bit_cast(__bf16, h);
}

// ---------------------------------------------------------------------------
// one-shot fp32 -> bf16 row-major copy
// ---------------------------------------------------------------------------
__global__ void pack_bf16(const float* __restrict__ in, __bf16* __restrict__ out,
                          long long n) {
  long long t = (long long)blockIdx.x * blockDim.x + threadIdx.x;
  if (t < n) out[t] = f2bf(in[t]);
}

// ---------------------------------------------------------------------------
// pack row-major B[K,Nc] (fp32) into WMMA lane layout:
//   Bp[c][n][g][e]  (c = K/32 chunk, g = lane half, e = 0..15)
//   lane L of the wave reads 16 contiguous bf16 at ((c*Nc + n)*2 + g)*16
// element e maps to K = 32c + 8g + e          (e < 8)
//                  K = 32c + 8g + 8 + e       (e >= 8)   [i.e. +16 .. +23 / +24..31]
// ---------------------------------------------------------------------------
__global__ void pack_B(const float* __restrict__ B, __bf16* __restrict__ Bp,
                       int K, int Nc) {
  long long t = (long long)blockIdx.x * blockDim.x + threadIdx.x;
  if (t >= (long long)K * Nc) return;
  int e  = (int)(t & 15);
  int g  = (int)((t >> 4) & 1);
  int n  = (int)((t >> 5) % Nc);
  int c  = (int)(t / ((long long)Nc * 32));
  int kk = 32 * c + 8 * g + e + (e >= 8 ? 8 : 0);
  Bp[t] = f2bf(B[(size_t)kk * Nc + n]);
}

// ---------------------------------------------------------------------------
// C[M x Nc](f32) = A[M x K](bf16 row-major) * Bp(packed bf16), WMMA 16x16x32.
// One wave -> one 16x16 tile. Requires M%16==0, K%32==0, Nc%16==0.
// ---------------------------------------------------------------------------
__global__ void gemm_wmma_bf16pk(const __bf16* __restrict__ A,
                                 const __bf16* __restrict__ Bp,
                                 float* __restrict__ C,
                                 int M, int K, int Nc) {
  const int lane = threadIdx.x & 31;
  const int wid  = threadIdx.x >> 5;
  const int m0   = blockIdx.x * 16;
  const int n0   = (blockIdx.y * (blockDim.x >> 5) + wid) * 16;
  if (m0 >= M || n0 >= Nc) return;          // uniform per wave -> EXEC stays full

  const int g    = lane >> 4;               // lane half
  const int mrow = m0 + (lane & 15);
  const int ncol = n0 + (lane & 15);

  const __bf16* arow = A  + (size_t)mrow * K + 8 * g;
  const __bf16* brow = Bp + ((size_t)ncol * 2 + g) * 16;
  const size_t  bstr = (size_t)Nc * 32;     // bf16 elements per k-chunk of Bp

  v8f acc = {};
  const int nch = K >> 5;
#pragma unroll 4
  for (int c = 0; c < nch; ++c) {
    v8bf alo = *(const v8bf*)(arow + 32 * c);        // K = 32c+8g   .. +7
    v8bf ahi = *(const v8bf*)(arow + 32 * c + 16);   // K = 32c+8g+16.. +23
    v16bf a  = __builtin_shufflevector(alo, ahi,
                 0,1,2,3,4,5,6,7,8,9,10,11,12,13,14,15);
    v16bf b  = *(const v16bf*)(brow + bstr * c);
    acc = __builtin_amdgcn_wmma_f32_16x16x32_bf16(false, a, false, b,
                                                  (short)0, acc, false, false);
  }
  const int rbase = m0 + (g << 3);
#pragma unroll
  for (int r = 0; r < 8; ++r)
    C[(size_t)(rbase + r) * Nc + ncol] = acc[r];
}

// ---------------------------------------------------------------------------
// attention scalars: as[n*H+h] = sum_c h[n,h,c]*a_src[h,c] (same for ad)
// ---------------------------------------------------------------------------
__global__ void att_scores(const float* __restrict__ h,
                           const float* __restrict__ a_src,
                           const float* __restrict__ a_dst,
                           float* __restrict__ as, float* __restrict__ ad,
                           long long NH, int H, int C) {
  long long t = (long long)blockIdx.x * blockDim.x + threadIdx.x;
  if (t >= NH) return;
  int hh = (int)(t % H);
  const float* hp = h + (size_t)t * C;      // n*H*C + hh*C == t*C
  float s = 0.f, d = 0.f;
  for (int c = 0; c < C; ++c) {
    float v = hp[c];
    s += v * a_src[hh * C + c];
    d += v * a_dst[hh * C + c];
  }
  as[t] = s;
  ad[t] = d;
}

__global__ void zero_f(float* __restrict__ p, long long n) {
  long long t = (long long)blockIdx.x * blockDim.x + threadIdx.x;
  if (t < n) p[t] = 0.f;
}

__global__ void init_md(float* __restrict__ m, float* __restrict__ den, long long n) {
  long long t = (long long)blockIdx.x * blockDim.x + threadIdx.x;
  if (t < n) { m[t] = -__builtin_inff(); den[t] = 0.f; }
}

__global__ void fix_m(float* __restrict__ m, long long n) {
  long long t = (long long)blockIdx.x * blockDim.x + threadIdx.x;
  if (t < n) { float v = m[t]; m[t] = __builtin_isfinite(v) ? v : 0.f; }
}

__device__ __forceinline__ void edge_sd(const long long* __restrict__ ei,
                                        long long E, long long eid,
                                        int& s, int& d) {
  if (eid < E) { s = (int)ei[eid]; d = (int)ei[E + eid]; }
  else         { s = d = (int)(eid - E); }          // self-loop tail
}

__device__ __forceinline__ float leaky(float x) { return x > 0.f ? x : 0.2f * x; }

__global__ void edge_max(const long long* __restrict__ ei, long long E, long long EE,
                         const float* __restrict__ as, const float* __restrict__ ad,
                         float* __restrict__ m, int H) {
  long long t = (long long)blockIdx.x * blockDim.x + threadIdx.x;
  if (t >= EE * H) return;
  long long eid = t / H; int hh = (int)(t % H);
  int s, d; edge_sd(ei, E, eid, s, d);
  float e = leaky(as[(size_t)s * H + hh] + ad[(size_t)d * H + hh]);
  __hip_atomic_fetch_max(&m[(size_t)d * H + hh], e,
                         __ATOMIC_RELAXED, __HIP_MEMORY_SCOPE_AGENT);
}

__global__ void edge_exp(const long long* __restrict__ ei, long long E, long long EE,
                         const float* __restrict__ as, const float* __restrict__ ad,
                         const float* __restrict__ m, float* __restrict__ den,
                         float* __restrict__ ex, int H) {
  long long t = (long long)blockIdx.x * blockDim.x + threadIdx.x;
  if (t >= EE * H) return;
  long long eid = t / H; int hh = (int)(t % H);
  int s, d; edge_sd(ei, E, eid, s, d);
  float e = leaky(as[(size_t)s * H + hh] + ad[(size_t)d * H + hh]);
  float x = __expf(e - m[(size_t)d * H + hh]);
  ex[t] = x;
  atomicAdd(&den[(size_t)d * H + hh], x);
}

__global__ void scatter_msg(const long long* __restrict__ ei, long long E, long long EE,
                            const float* __restrict__ ex, const float* __restrict__ den,
                            const float* __restrict__ feat, float* __restrict__ out,
                            int H, int C, int outStride, int outColOff) {
  long long t = (long long)blockIdx.x * blockDim.x + threadIdx.x;
  const int HC = H * C;
  if (t >= EE * HC) return;
  long long eid = t / HC;
  int r  = (int)(t % HC);
  int hh = r / C;
  int s, d; edge_sd(ei, E, eid, s, d);
  float alpha = ex[eid * H + hh] / (den[(size_t)d * H + hh] + 1e-16f);
  atomicAdd(&out[(size_t)d * outStride + outColOff + r],
            feat[(size_t)s * HC + r] * alpha);
}

// bias + ELU, emitting the bf16 activation consumed by the layer-2 GEMM
__global__ void postact_elu(const float* __restrict__ x2,
                            const float* __restrict__ bias,
                            __bf16* __restrict__ x2bf,
                            long long n, int F) {
  long long t = (long long)blockIdx.x * blockDim.x + threadIdx.x;
  if (t >= n) return;
  int j = (int)(t % F);
  float v = x2[t] + bias[j];
  v = v > 0.f ? v : (__expf(v) - 1.f);
  x2bf[t] = f2bf(v);
}

__global__ void bias_add(float* __restrict__ out, const float* __restrict__ bias,
                         long long n, int C) {
  long long t = (long long)blockIdx.x * blockDim.x + threadIdx.x;
  if (t >= n) return;
  out[t] += bias[(int)(t % C)];
}

// ---------------------------------------------------------------------------
extern "C" void kernel_launch(void* const* d_in, const int* in_sizes, int n_in,
                              void* d_out, int out_size, void* d_ws, size_t ws_size,
                              hipStream_t stream) {
  const float*     x_list = (const float*)d_in[0];     // [2, N, 128]
  const long long* ei     = (const long long*)d_in[1]; // [2, E] int64
  const float*     W1     = (const float*)d_in[2];     // [2, 128, 256]
  const float*     aS1    = (const float*)d_in[3];     // [2, 4, 64]
  const float*     aD1    = (const float*)d_in[4];     // [2, 4, 64]
  const float*     b1     = (const float*)d_in[5];     // [2, 256] (contig = [512])
  const float*     W2     = (const float*)d_in[6];     // [512, 16]
  const float*     aS2    = (const float*)d_in[7];     // [1, 16]
  const float*     aD2    = (const float*)d_in[8];     // [1, 16]
  const float*     b2     = (const float*)d_in[9];     // [16]
  float*           out    = (float*)d_out;             // [N, 16]

  const int F_in = 128, HC1 = 256, H1 = 4, C1 = 64, F2 = 512, H2 = 1, C2 = 16;
  const long long N  = (long long)in_sizes[0] / (2 * F_in);
  const long long E  = (long long)in_sizes[1] / 2;
  const long long EE = E + N;                 // + self loops

  // ---- workspace layout ----
  float* ws  = (float*)d_ws;
  float* h1  = ws;                 // N*256  (per-conv gemm out, reused)
  float* x2  = h1  + N * HC1;      // N*512  layer-1 concat accumulator (f32)
  float* h2  = x2  + N * F2;       // N*16
  float* as  = h2  + N * C2;       // N*4
  float* ad  = as  + N * H1;       // N*4
  float* m   = ad  + N * H1;       // N*4
  float* den = m   + N * H1;       // N*4
  float* ex  = den + N * H1;       // EE*4
  // bf16 section (32B aligned: all float offsets above are multiples of 8)
  long long exn = ((EE * H1 + 15) / 16) * 16;
  __bf16* x2bf = (__bf16*)(ex + exn);          // N*512 bf16
  __bf16* abf  = x2bf;                         // alias: N*128 bf16, dead before
                                               // x2bf is written (postact)
  __bf16* wp   = x2bf + N * F2;                // weight pack, max 128*256 bf16

  auto nb = [](long long n_, int b) { return (unsigned)((n_ + b - 1) / b); };

  zero_f<<<nb(N * F2, 256), 256, 0, stream>>>(x2, N * F2);

  for (int k = 0; k < 2; ++k) {
    const float* A = x_list + (size_t)k * N * F_in;
    const float* B = W1 + (size_t)k * F_in * HC1;
    pack_bf16<<<nb(N * F_in, 256), 256, 0, stream>>>(A, abf, N * F_in);
    pack_B<<<nb((long long)F_in * HC1, 256), 256, 0, stream>>>(B, wp, F_in, HC1);
    dim3 g((unsigned)(N / 16), HC1 / 64);                  // 4 waves/block
    gemm_wmma_bf16pk<<<g, 128, 0, stream>>>(abf, wp, h1, (int)N, F_in, HC1);

    att_scores<<<nb(N * H1, 256), 256, 0, stream>>>(h1, aS1 + k * H1 * C1,
                                                    aD1 + k * H1 * C1,
                                                    as, ad, N * H1, H1, C1);
    init_md<<<nb(N * H1, 256), 256, 0, stream>>>(m, den, N * H1);
    edge_max<<<nb(EE * H1, 256), 256, 0, stream>>>(ei, E, EE, as, ad, m, H1);
    fix_m<<<nb(N * H1, 256), 256, 0, stream>>>(m, N * H1);
    edge_exp<<<nb(EE * H1, 256), 256, 0, stream>>>(ei, E, EE, as, ad, m, den, ex, H1);
    scatter_msg<<<nb(EE * H1 * C1, 256), 256, 0, stream>>>(ei, E, EE, ex, den, h1,
                                                           x2, H1, C1, F2, k * HC1);
  }
  postact_elu<<<nb(N * F2, 256), 256, 0, stream>>>(x2, b1, x2bf, N * F2, F2);

  // ---- layer 2 ----
  pack_B<<<nb((long long)F2 * C2, 256), 256, 0, stream>>>(W2, wp, F2, C2);
  {
    dim3 g((unsigned)(N / 16), 1);                          // Nc == 16 -> 1 wave/block
    gemm_wmma_bf16pk<<<g, 32, 0, stream>>>(x2bf, wp, h2, (int)N, F2, C2);
  }
  att_scores<<<nb(N * H2, 256), 256, 0, stream>>>(h2, aS2, aD2, as, ad, N * H2, H2, C2);
  init_md<<<nb(N * H2, 256), 256, 0, stream>>>(m, den, N * H2);
  edge_max<<<nb(EE * H2, 256), 256, 0, stream>>>(ei, E, EE, as, ad, m, H2);
  fix_m<<<nb(N * H2, 256), 256, 0, stream>>>(m, N * H2);
  edge_exp<<<nb(EE * H2, 256), 256, 0, stream>>>(ei, E, EE, as, ad, m, den, ex, H2);

  zero_f<<<nb(N * C2, 256), 256, 0, stream>>>(out, N * C2);
  scatter_msg<<<nb(EE * H2 * C2, 256), 256, 0, stream>>>(ei, E, EE, ex, den, h2,
                                                         out, H2, C2, C2, 0);
  bias_add<<<nb(N * C2, 256), 256, 0, stream>>>(out, b2, N * C2, C2);
}
